// MultiHeadAttentionKVCache_40827959116090
// MI455X (gfx1250) — matrix operand order repaired and checked by reading, not compiled, not run on hardware
//
#include <hip/hip_runtime.h>

#define N_STATE  1024
#define N_HEAD   16
#define HEAD_DIM 64
#define BATCH    32
#define CACHE    4096
#define SCALE2   0.125f   // (64^-0.25)^2 = 1/8

typedef __attribute__((ext_vector_type(2)))  float    v2f;
typedef __attribute__((ext_vector_type(4)))  float    v4f;
typedef __attribute__((ext_vector_type(8)))  float    v8f;
typedef __attribute__((ext_vector_type(16))) _Float16 v16h;

#if defined(__has_builtin)
#if __has_builtin(__builtin_amdgcn_wmma_f32_16x16x4_f32)
#define HAVE_WMMA_F32X4 1
#endif
#endif

// ---------------------------------------------------------------------------
// Y[32,1024] = X[32,1024] @ W^T (+ bias).  W is [out,in] row-major.
// One block = one 16-wide N tile; wave0 -> rows 0..15, wave1 -> rows 16..31.
// ---------------------------------------------------------------------------
__global__ __launch_bounds__(64) void proj_kernel(
    const float* __restrict__ X,
    const float* __restrict__ W,
    const float* __restrict__ bias,
    float* __restrict__ Y)
{
    const int nbase = blockIdx.x * 16;
    const int wave  = threadIdx.x >> 5;
    const int lane  = threadIdx.x & 31;
    const int mbase = wave * 16;
    const int n     = lane & 15;
    const bool hi   = lane >= 16;

    v8f c = {};

#if HAVE_WMMA_F32X4
    // A 16x4 f32: lanes 0-15 hold K=k0,k0+1; lanes 16-31 hold K=k0+2,k0+3.
    // B  4x16 f32: same K split, N = lane%16 (column of W^T == row of W).
    const float* xrow = X + (size_t)(mbase + n) * N_STATE + (hi ? 2 : 0);
    const float* wrow = W + (size_t)(nbase + n) * N_STATE + (hi ? 2 : 0);
#pragma unroll 4
    for (int k = 0; k < N_STATE; k += 4) {
        v2f a = *(const v2f*)(xrow + k);
        v2f b = *(const v2f*)(wrow + k);
        c = __builtin_amdgcn_wmma_f32_16x16x4_f32(
                false, a, false, b, (short)0, c, false, false);
    }
#else
    // f16 fallback, K-step 32 (fp32 accumulate).
    const float* xr0 = X + (size_t)(mbase + n) * N_STATE;
    const float* wr0 = W + (size_t)(nbase + n) * N_STATE + (hi ? 16 : 0);
    const int a1 = hi ? 16 : 0, a2 = hi ? 24 : 8;
#pragma unroll 2
    for (int k = 0; k < N_STATE; k += 32) {
        v16h a, b;
#pragma unroll
        for (int i = 0; i < 8; ++i) {
            a[i]     = (_Float16)xr0[k + a1 + i];
            a[8 + i] = (_Float16)xr0[k + a2 + i];
        }
#pragma unroll
        for (int i = 0; i < 16; ++i) b[i] = (_Float16)wr0[k + i];
        c = __builtin_amdgcn_wmma_f32_16x16x32_f16(
                false, a, false, b, (short)0, c, false, false);
    }
#endif

    const float bn = bias ? bias[nbase + n] : 0.0f;
#pragma unroll
    for (int r = 0; r < 8; ++r) {
        const int m = mbase + r + (hi ? 8 : 0);
        Y[(size_t)m * N_STATE + nbase + n] = c[r] + bn;
    }
}

// ---------------------------------------------------------------------------
// Flash-decode attention. One block per (batch, head); 8 waves stream the
// 4096 cached rows in 16-key blocks; scores via WMMA (q replicated in A).
// new_key/new_value row handled at the merge.  ctx[b, h*64 + d] written.
// ---------------------------------------------------------------------------
__global__ __launch_bounds__(256) void attn_kernel(
    const float* __restrict__ q,          // [32,1024], bias included
    const float* __restrict__ past_key,   // [32,4096,1024]
    const float* __restrict__ past_value, // [32,4096,1024]
    const float* __restrict__ new_key,    // [32,1024]
    const float* __restrict__ new_value,  // [32,1024]
    const float* __restrict__ mask,       // [4097]
    float* __restrict__ ctx)              // [32,1024]
{
    const int bh   = blockIdx.x;
    const int b    = bh >> 4;
    const int h    = bh & 15;
    const int tid  = threadIdx.x;
    const int wave = tid >> 5;
    const int lane = tid & 31;
    const int n    = lane & 15;
    const bool hi  = lane >= 16;
    const int d2   = 2 * lane;            // this lane's two output dims

    const float* qh = q          + (size_t)b * N_STATE + h * HEAD_DIM;
    const float* Kb = past_key   + (size_t)b * CACHE * N_STATE + h * HEAD_DIM;
    const float* Vb = past_value + (size_t)b * CACHE * N_STATE + h * HEAD_DIM;

    // A fragments: every row = q * SCALE^2 (rows replicated -> any D row ok).
    v16h af0, af1;
    {
        const int b1 = hi ? 16 : 0, b2 = hi ? 24 : 8;
#pragma unroll
        for (int i = 0; i < 8; ++i) {
            af0[i]     = (_Float16)(qh[b1 + i] * SCALE2);
            af0[8 + i] = (_Float16)(qh[b2 + i] * SCALE2);
            af1[i]     = (_Float16)(qh[32 + b1 + i] * SCALE2);
            af1[8 + i] = (_Float16)(qh[32 + b2 + i] * SCALE2);
        }
    }

    float m_run = -3.0e38f;
    float l_run = 0.0f;
    float acc0  = 0.0f, acc1 = 0.0f;

    for (int blk = wave; blk < CACHE / 16; blk += 8) {
        const int j0 = blk * 16;
        // B frag: this lane supplies key row (j0+n), head dims [o..o+15] and
        // [32+o..32+o+15] with o = hi?16:0  (contiguous -> b128 loads).
        const float* krow = Kb + (size_t)(j0 + n) * N_STATE + (hi ? 16 : 0);
        const float* vrow = Vb + (size_t)j0 * N_STATE + d2;

        if (blk + 8 < CACHE / 16) {
            __builtin_prefetch(krow + 128 * N_STATE, 0, 0);
            __builtin_prefetch(vrow + 128 * N_STATE, 0, 0);
        }

        v16h bf0, bf1;
#pragma unroll
        for (int i = 0; i < 4; ++i) {
            v4f k0 = *(const v4f*)(krow + 4 * i);
            v4f k1 = *(const v4f*)(krow + 32 + 4 * i);
            bf0[4*i+0] = (_Float16)k0.x; bf0[4*i+1] = (_Float16)k0.y;
            bf0[4*i+2] = (_Float16)k0.z; bf0[4*i+3] = (_Float16)k0.w;
            bf1[4*i+0] = (_Float16)k1.x; bf1[4*i+1] = (_Float16)k1.y;
            bf1[4*i+2] = (_Float16)k1.z; bf1[4*i+3] = (_Float16)k1.w;
        }

        v8f c = {};
        c = __builtin_amdgcn_wmma_f32_16x16x32_f16(
                false, af0, false, bf0, (short)0, c, false, false);
        c = __builtin_amdgcn_wmma_f32_16x16x32_f16(
                false, af1, false, bf1, (short)0, c, false, false);

        // c[0] = score of key j0+n (replicated across half-waves).
        const float s = c[0] + mask[j0 + n];

        float bm = s;
#pragma unroll
        for (int off = 1; off <= 16; off <<= 1)
            bm = fmaxf(bm, __shfl_xor(bm, off, 32));

        const float m_new = fmaxf(m_run, bm);
        const float r     = __expf(m_run - m_new);
        const float e     = __expf(s - m_new);

        float es = e;
#pragma unroll
        for (int off = 1; off <= 8; off <<= 1)
            es += __shfl_xor(es, off, 32);

        l_run = l_run * r + es;
        m_run = m_new;
        acc0 *= r;
        acc1 *= r;

#pragma unroll
        for (int jj = 0; jj < 16; ++jj) {
            const float w = __shfl(e, jj, 32);
            v2f v = *(const v2f*)(vrow + (size_t)jj * N_STATE);
            acc0 += w * v.x;
            acc1 += w * v.y;
        }
    }

    __shared__ float s_m[8], s_l[8];
    __shared__ float s_acc[8][HEAD_DIM];
    if (lane == 0) { s_m[wave] = m_run; s_l[wave] = l_run; }
    s_acc[wave][d2]     = acc0;
    s_acc[wave][d2 + 1] = acc1;
    __syncthreads();

    if (wave == 0) {
        // contribution of the freshly projected token
        const float* nk = new_key   + (size_t)b * N_STATE + h * HEAD_DIM;
        const float* nv = new_value + (size_t)b * N_STATE + h * HEAD_DIM;
        float p = qh[d2] * nk[d2] + qh[d2 + 1] * nk[d2 + 1];
#pragma unroll
        for (int off = 1; off <= 16; off <<= 1)
            p += __shfl_xor(p, off, 32);
        const float s_last = p * SCALE2 + mask[CACHE];

        float mstar = s_last;
#pragma unroll
        for (int w = 0; w < 8; ++w) mstar = fmaxf(mstar, s_m[w]);

        float denom = __expf(s_last - mstar);
        float num0  = denom * nv[d2];
        float num1  = denom * nv[d2 + 1];
#pragma unroll
        for (int w = 0; w < 8; ++w) {
            const float f = __expf(s_m[w] - mstar);
            denom += f * s_l[w];
            num0  += f * s_acc[w][d2];
            num1  += f * s_acc[w][d2 + 1];
        }
        const float inv = 1.0f / denom;
        float* o = ctx + (size_t)b * N_STATE + h * HEAD_DIM;
        o[d2]     = num0 * inv;
        o[d2 + 1] = num1 * inv;
    }
}

// ---------------------------------------------------------------------------
extern "C" void kernel_launch(void* const* d_in, const int* in_sizes, int n_in,
                              void* d_out, int out_size, void* d_ws, size_t ws_size,
                              hipStream_t stream)
{
    const float* hs        = (const float*)d_in[0];
    const float* past_key  = (const float*)d_in[1];
    const float* past_val  = (const float*)d_in[2];
    const float* attn_mask = (const float*)d_in[3];
    const float* Wq        = (const float*)d_in[4];
    const float* bq        = (const float*)d_in[5];
    const float* Wk        = (const float*)d_in[6];
    const float* Wv        = (const float*)d_in[7];
    const float* bv        = (const float*)d_in[8];
    const float* Wo        = (const float*)d_in[9];
    const float* bo        = (const float*)d_in[10];

    float* out      = (float*)d_out;
    float* out_attn = out;                        // [32,1024]
    float* out_key  = out + BATCH * N_STATE;      // [32,1024]
    float* out_val  = out + 2 * BATCH * N_STATE;  // [32,1024]

    float* ws_q   = (float*)d_ws;                 // 128 KB
    float* ws_ctx = ws_q + BATCH * N_STATE;       // 128 KB

    proj_kernel<<<dim3(N_STATE / 16), dim3(64), 0, stream>>>(hs, Wq, bq, ws_q);
    proj_kernel<<<dim3(N_STATE / 16), dim3(64), 0, stream>>>(hs, Wk, nullptr, out_key);
    proj_kernel<<<dim3(N_STATE / 16), dim3(64), 0, stream>>>(hs, Wv, bv, out_val);

    attn_kernel<<<dim3(BATCH * N_HEAD), dim3(256), 0, stream>>>(
        ws_q, past_key, past_val, out_key, out_val, attn_mask, ws_ctx);

    proj_kernel<<<dim3(N_STATE / 16), dim3(64), 0, stream>>>(ws_ctx, Wo, bo, out_attn);
}
